// GAT_38817914421446
// MI455X (gfx1250) — compile-verified, matrix operand order checked by reading, not run on hardware
//
#include <hip/hip_runtime.h>
#include <math.h>

// ---------------- problem constants (from reference setup_inputs) -----------
#define NN    2048        // nodes
#define FIN   33          // input features
#define D1    64          // hidden = H*C
#define HH    8           // heads
#define CC    8           // channels/head
#define NE    65536       // edges
#define NET   (NE + NN)   // edges + self loops
#define MSZ   1024        // sourceSize == targetSize
#define ZN    1025        // Sinkhorn matrix side (m+1)
#define ZP    1028        // padded row stride (float4-aligned)
#define PADV  (-1e30f)    // pad value: exp(PADV - m) == 0, never wins max
#define SINK_ITERS 1000
#define SINK_BLOCKS 128
#define LOG2048F 7.6246189861593985f   // log(m+n)
#define LOG1024F 6.9314718055994531f   // log(1024)

typedef __bf16 bf16_t;
typedef __attribute__((ext_vector_type(16))) __bf16 v16bf;
typedef __attribute__((ext_vector_type(8)))  float  v8f;

// ---------------- helpers ---------------------------------------------------
__device__ __forceinline__ void atomicMaxF(float* a, float v) {
  // IEEE ordering trick: valid for all finite floats with -inf init.
  if (v >= 0.f) atomicMax((int*)a, __float_as_int(v));
  else          atomicMin((unsigned int*)a, __float_as_uint(v));
}

__device__ __forceinline__ unsigned pk2bf(float x, float y) {
  unsigned short lo = __builtin_bit_cast(unsigned short, (bf16_t)x);
  unsigned short hi = __builtin_bit_cast(unsigned short, (bf16_t)y);
  return (unsigned)lo | ((unsigned)hi << 16);
}

// ---------------- WMMA GEMM: C(MxN) = A(MxK=64) * W(NxK=64)^T ---------------
// One wave per 16x16 C tile; K fixed at 64 (two 16x16x32 bf16 WMMAs).
// float4 global staging -> packed bf16 LDS; fragment layouts per ISA 7.12.2:
//   A 16x32 bf16 : lane L<16 -> row M=L,   K = h + 8*(h>=8)
//                  lane L>=16 -> row M=L-16,K = h + 8 + 8*(h>=8)
//   B staged as [n][k], same striping with N in place of M
//   C/D 16x16 f32: N = lane&15, M = vgpr + 8*(lane>=16)
__global__ void __launch_bounds__(32)
k_gemm64(const float* __restrict__ A, const float* __restrict__ W,
         float* __restrict__ C, int ldc) {
  const int tm = blockIdx.x * 16, tn = blockIdx.y * 16;
  const int lane = threadIdx.x;
  __shared__ unsigned AsU[256];   // 512 bf16
  __shared__ unsigned BsU[256];
  const bf16_t* As = (const bf16_t*)AsU;
  const bf16_t* Bs = (const bf16_t*)BsU;
  v8f acc = {};
#pragma unroll
  for (int k0 = 0; k0 < 64; k0 += 32) {
#pragma unroll
    for (int i = 0; i < 4; ++i) {
      int f = (lane + i * 32) * 4;            // flat float idx in 16x32 tile
      int r = f >> 5, c = f & 31;
      float4 a4 = *(const float4*)(A + (size_t)(tm + r) * 64 + k0 + c);
      float4 b4 = *(const float4*)(W + (size_t)(tn + r) * 64 + k0 + c);
      AsU[f / 2]     = pk2bf(a4.x, a4.y);
      AsU[f / 2 + 1] = pk2bf(a4.z, a4.w);
      BsU[f / 2]     = pk2bf(b4.x, b4.y);
      BsU[f / 2 + 1] = pk2bf(b4.z, b4.w);
    }
    __syncthreads();
    v16bf af, bfv;
    const int row  = lane & 15;
    const int koff = (lane >> 4) * 8;
#pragma unroll
    for (int h = 0; h < 16; ++h) {
      int k = koff + h + ((h >= 8) ? 8 : 0);
      af[h]  = As[row * 32 + k];
      bfv[h] = Bs[row * 32 + k];
    }
    acc = __builtin_amdgcn_wmma_f32_16x16x32_bf16(false, af, false, bfv,
                                                  (short)0, acc, false, false);
    __syncthreads();
  }
  const int n  = tn + (lane & 15);
  const int mb = tm + ((lane >> 4) * 8);
#pragma unroll
  for (int r = 0; r < 8; ++r)
    C[(size_t)(mb + r) * ldc + n] = acc[r];
}

// zero-pad a (rows x FIN) row-major matrix into (rows x 64)
__global__ void k_pad33(const float* __restrict__ in, float* __restrict__ out,
                        int rows) {
  int idx = blockIdx.x * blockDim.x + threadIdx.x;
  if (idx >= rows * 64) return;
  int r = idx >> 6, c = idx & 63;
  out[idx] = (c < FIN) ? in[r * FIN + c] : 0.f;
}

// ---------------- GATv2 edge kernels ----------------------------------------
__global__ void k_init_layer(float* mx, float* denom) {
  int i = blockIdx.x * blockDim.x + threadIdx.x;
  if (i < NN * HH) { mx[i] = -INFINITY; denom[i] = 0.f; }
}

__global__ void k_init_out(float* out, const float* __restrict__ bias) {
  int i = blockIdx.x * blockDim.x + threadIdx.x;
  if (i < NN * D1) out[i] = bias[i & (D1 - 1)];
}

__global__ void k_edge_logits(const int* __restrict__ ei,
                              const float* __restrict__ xl,
                              const float* __restrict__ xr,
                              const float* __restrict__ att,
                              float* __restrict__ logits,
                              float* __restrict__ mx) {
  int e = blockIdx.x * blockDim.x + threadIdx.x;
  if (e >= NET) return;
  int src = (e < NE) ? ei[e]      : (e - NE);
  int dst = (e < NE) ? ei[NE + e] : (e - NE);
  const float4* pl = (const float4*)(xl + (size_t)src * D1);
  const float4* pr = (const float4*)(xr + (size_t)dst * D1);
#pragma unroll
  for (int h = 0; h < HH; ++h) {
    float4 l0 = pl[h * 2], l1 = pl[h * 2 + 1];
    float4 r0 = pr[h * 2], r1 = pr[h * 2 + 1];
    float sv[8] = { l0.x + r0.x, l0.y + r0.y, l0.z + r0.z, l0.w + r0.w,
                    l1.x + r1.x, l1.y + r1.y, l1.z + r1.z, l1.w + r1.w };
    float acc = 0.f;
#pragma unroll
    for (int c = 0; c < CC; ++c) {
      float s = sv[c];
      s = (s >= 0.f) ? s : 0.2f * s;            // leaky_relu(0.2)
      acc += s * att[h * CC + c];
    }
    logits[(size_t)e * HH + h] = acc;
    atomicMaxF(&mx[dst * HH + h], acc);
  }
}

__global__ void k_fix_mx(float* mx) {
  int i = blockIdx.x * blockDim.x + threadIdx.x;
  if (i < NN * HH && mx[i] == -INFINITY) mx[i] = 0.f;
}

__global__ void k_edge_exp(const int* __restrict__ ei,
                           float* __restrict__ logits,
                           const float* __restrict__ mx,
                           float* __restrict__ denom) {
  int e = blockIdx.x * blockDim.x + threadIdx.x;
  if (e >= NET) return;
  int dst = (e < NE) ? ei[NE + e] : (e - NE);
#pragma unroll
  for (int h = 0; h < HH; ++h) {
    float a = __expf(logits[(size_t)e * HH + h] - mx[dst * HH + h]);
    logits[(size_t)e * HH + h] = a;             // reuse buffer for a
    atomicAdd(&denom[dst * HH + h], a);
  }
}

__global__ void k_aggregate(const int* __restrict__ ei,
                            const float* __restrict__ a,
                            const float* __restrict__ denom,
                            const float* __restrict__ xl,
                            float* __restrict__ out) {
  int e = blockIdx.x * blockDim.x + threadIdx.x;
  if (e >= NET) return;
  int src = (e < NE) ? ei[e]      : (e - NE);
  int dst = (e < NE) ? ei[NE + e] : (e - NE);
  const float4* pl = (const float4*)(xl + (size_t)src * D1);
  float* po = out + (size_t)dst * D1;
#pragma unroll
  for (int h = 0; h < HH; ++h) {
    float alpha = a[(size_t)e * HH + h] / (denom[dst * HH + h] + 1e-16f);
    float4 f0 = pl[h * 2], f1 = pl[h * 2 + 1];
    atomicAdd(&po[h * CC + 0], f0.x * alpha);
    atomicAdd(&po[h * CC + 1], f0.y * alpha);
    atomicAdd(&po[h * CC + 2], f0.z * alpha);
    atomicAdd(&po[h * CC + 3], f0.w * alpha);
    atomicAdd(&po[h * CC + 4], f1.x * alpha);
    atomicAdd(&po[h * CC + 5], f1.y * alpha);
    atomicAdd(&po[h * CC + 6], f1.z * alpha);
    atomicAdd(&po[h * CC + 7], f1.w * alpha);
  }
}

// ---------------- row ops: log_softmax, L2 norm -----------------------------
__global__ void k_logsoftmax(float* __restrict__ h) {
  __shared__ float red[64];
  int row = blockIdx.x, t = threadIdx.x;
  float x = h[(size_t)row * D1 + t];
  red[t] = x; __syncthreads();
  for (int off = 32; off > 0; off >>= 1) {
    if (t < off) red[t] = fmaxf(red[t], red[t + off]);
    __syncthreads();
  }
  float m = red[0]; __syncthreads();
  red[t] = __expf(x - m); __syncthreads();
  for (int off = 32; off > 0; off >>= 1) {
    if (t < off) red[t] += red[t + off];
    __syncthreads();
  }
  float ls = __logf(red[0]);
  h[(size_t)row * D1 + t] = x - m - ls;
}

__global__ void k_rownorm(const float* __restrict__ h, float* __restrict__ norms) {
  __shared__ float red[64];
  int row = blockIdx.x, t = threadIdx.x;
  float x = h[(size_t)row * D1 + t];
  red[t] = x * x; __syncthreads();
  for (int off = 32; off > 0; off >>= 1) {
    if (t < off) red[t] += red[t + off];
    __syncthreads();
  }
  if (t == 0) norms[row] = sqrtf(red[0]);
}

// ---------------- Z0 assembly (padded stride ZP) ----------------------------
// interior: cosine-normalized scores; Sinkhorn borders: dust_bin;
// pad lanes: -1e30 (zero weight in LSE). Also builds Z0^T for coalesced cols.
__global__ void k_finalizeZ0(float* __restrict__ Z0, float* __restrict__ Z0T,
                             const float* __restrict__ norms,
                             const float* __restrict__ dust) {
  int j = blockIdx.x * blockDim.x + threadIdx.x;
  int i = blockIdx.y;
  if (j >= ZP) return;
  float z;
  if (i < MSZ && j < MSZ)
    z = Z0[(size_t)i * ZP + j] / (norms[i] * norms[MSZ + j]);
  else if (i < ZN && j < ZN)
    z = dust[0];
  else
    z = PADV;
  Z0[(size_t)i * ZP + j]  = z;
  Z0T[(size_t)j * ZP + i] = z;
}

__global__ void k_init_sink(float* u, float* v, unsigned* bar) {
  int i = blockIdx.x * blockDim.x + threadIdx.x;
  if (i < ZP) { u[i] = 0.f; v[i] = 0.f; }   // pad entries stay 0 forever
  if (i < 8) bar[i] = 0u;
}

// ---------------- persistent Sinkhorn ---------------------------------------
__device__ __forceinline__ void lse_block_reduce(float* sm, float* ss, int t) {
  __syncthreads();
  for (int off = 128; off > 0; off >>= 1) {
    if (t < off) {
      float m1 = sm[t],       s1 = ss[t];
      float m2 = sm[t + off], s2 = ss[t + off];
      float mo = fmaxf(m1, m2);
      ss[t] = s1 * __expf(m1 - mo) + s2 * __expf(m2 - mo);
      sm[t] = mo;
    }
    __syncthreads();
  }
}

__device__ __forceinline__ void grid_sync(unsigned* bar) {
  __syncthreads();
  __threadfence();                       // release
  if (threadIdx.x == 0) {
    volatile unsigned* vgen = bar + 1;
    unsigned gen = *vgen;
    if (atomicAdd(bar, 1u) == gridDim.x - 1) {
      atomicExch(bar, 0u);               // reset before publishing
      __threadfence();
      atomicAdd((unsigned*)(bar + 1), 1u);
    } else {
      while (*vgen == gen) __builtin_amdgcn_s_sleep(2);
    }
  }
  __syncthreads();
  __threadfence();                       // acquire
}

#define LSE_UPD(val)                                         \
  do {                                                       \
    float _v = (val);                                        \
    if (_v > m) { s = s * __expf(m - _v) + 1.f; m = _v; }    \
    else        { s += __expf(_v - m); }                     \
  } while (0)

__global__ void __launch_bounds__(256)
k_sinkhorn(const float* __restrict__ Z0, const float* __restrict__ Z0T,
           float* __restrict__ u, float* __restrict__ v, unsigned* bar) {
  __shared__ float sm[256], ss[256];
  const int t = threadIdx.x;
  const float normc = -LOG2048F;
  const float4* v4 = (const float4*)v;
  const float4* u4 = (const float4*)u;
  for (int it = 0; it < SINK_ITERS; ++it) {
    // u = log_mu - LSE_j(Z0[i,j] + v[j])   (row pass; float4, L2-resident)
    for (int row = blockIdx.x; row < ZN; row += gridDim.x) {
      const float4* zr = (const float4*)(Z0 + (size_t)row * ZP);
      float m = -INFINITY, s = 0.f;
      for (int j = t; j < ZP / 4; j += 256) {
        float4 z = zr[j], w = v4[j];
        LSE_UPD(z.x + w.x); LSE_UPD(z.y + w.y);
        LSE_UPD(z.z + w.z); LSE_UPD(z.w + w.w);
      }
      sm[t] = m; ss[t] = s;
      lse_block_reduce(sm, ss, t);
      if (t == 0) {
        float lmu = normc + ((row == ZN - 1) ? LOG1024F : 0.f);
        u[row] = lmu - (sm[0] + __logf(ss[0]));
      }
      __syncthreads();
    }
    grid_sync(bar);
    // v = log_nu - LSE_i(Z0[i,j] + u[i])   (col pass via Z0^T; coalesced)
    for (int col = blockIdx.x; col < ZN; col += gridDim.x) {
      const float4* zc = (const float4*)(Z0T + (size_t)col * ZP);
      float m = -INFINITY, s = 0.f;
      for (int i = t; i < ZP / 4; i += 256) {
        float4 z = zc[i], w = u4[i];
        LSE_UPD(z.x + w.x); LSE_UPD(z.y + w.y);
        LSE_UPD(z.z + w.z); LSE_UPD(z.w + w.w);
      }
      sm[t] = m; ss[t] = s;
      lse_block_reduce(sm, ss, t);
      if (t == 0) {
        float lnu = normc + ((col == ZN - 1) ? LOG1024F : 0.f);
        v[col] = lnu - (sm[0] + __logf(ss[0]));
      }
      __syncthreads();
    }
    grid_sync(bar);
  }
}

// ---------------- outputs ----------------------------------------------------
__global__ void k_writeZ(const float* __restrict__ Z0,
                         const float* __restrict__ u,
                         const float* __restrict__ v,
                         float* __restrict__ out) {
  size_t idx = (size_t)blockIdx.x * blockDim.x + threadIdx.x;
  if (idx >= (size_t)ZN * ZN) return;
  int i = (int)(idx / ZN), j = (int)(idx % ZN);
  out[idx] = Z0[(size_t)i * ZP + j] + u[i] + v[j] + LOG2048F;  // -norm
}

__global__ void k_argmax_rows(const float* __restrict__ Z0,
                              const float* __restrict__ v,
                              int* __restrict__ ind) {
  __shared__ float sv[256]; __shared__ int si[256];
  int i = blockIdx.x, t = threadIdx.x;
  float best = -INFINITY; int bi = MSZ;
  for (int j = t; j < MSZ; j += 256) {          // +u[i]-norm const: ignore
    float val = Z0[(size_t)i * ZP + j] + v[j];
    if (val > best) { best = val; bi = j; }     // strict > keeps first idx
  }
  sv[t] = best; si[t] = bi; __syncthreads();
  for (int off = 128; off > 0; off >>= 1) {
    if (t < off) {
      float v2 = sv[t + off]; int i2 = si[t + off];
      if (v2 > sv[t] || (v2 == sv[t] && i2 < si[t])) { sv[t] = v2; si[t] = i2; }
    }
    __syncthreads();
  }
  if (t == 0) ind[i] = si[0];
}

__global__ void k_argmax_cols(const float* __restrict__ Z0T,
                              const float* __restrict__ u,
                              int* __restrict__ ind) {
  __shared__ float sv[256]; __shared__ int si[256];
  int j = blockIdx.x, t = threadIdx.x;
  float best = -INFINITY; int bi = MSZ;
  for (int i = t; i < MSZ; i += 256) {
    float val = Z0T[(size_t)j * ZP + i] + u[i];
    if (val > best) { best = val; bi = i; }
  }
  sv[t] = best; si[t] = bi; __syncthreads();
  for (int off = 128; off > 0; off >>= 1) {
    if (t < off) {
      float v2 = sv[t + off]; int i2 = si[t + off];
      if (v2 > sv[t] || (v2 == sv[t] && i2 < si[t])) { sv[t] = v2; si[t] = i2; }
    }
    __syncthreads();
  }
  if (t == 0) ind[j] = si[0];
}

__global__ void k_mutual0(const int* __restrict__ ind0,
                          const int* __restrict__ ind1,
                          int* __restrict__ mut0, float* __restrict__ i0out) {
  int i = blockIdx.x * blockDim.x + threadIdx.x;
  if (i >= MSZ) return;
  int j = ind0[i];
  int m = (ind1[j] == i) ? 1 : 0;
  mut0[i] = m;
  i0out[i] = m ? (float)j : -1.f;
}

__global__ void k_i1(const int* __restrict__ ind0,
                     const int* __restrict__ ind1,
                     const int* __restrict__ mut0, float* __restrict__ i1out) {
  int j = blockIdx.x * blockDim.x + threadIdx.x;
  if (j >= MSZ) return;
  int i = ind1[j];
  int valid = (ind0[i] == j) && mut0[i];
  i1out[j] = valid ? (float)i : -1.f;
}

// ---------------- launcher ---------------------------------------------------
extern "C" void kernel_launch(void* const* d_in, const int* in_sizes, int n_in,
                              void* d_out, int out_size, void* d_ws, size_t ws_size,
                              hipStream_t stream) {
  (void)in_sizes; (void)n_in; (void)out_size; (void)ws_size;
  const float* x    = (const float*)d_in[0];
  const int*   ei1  = (const int*)d_in[1];
  const int*   ei2  = (const int*)d_in[2];
  const float* Wl1  = (const float*)d_in[3];
  const float* Wr1  = (const float*)d_in[4];
  const float* att1 = (const float*)d_in[5];
  const float* b1   = (const float*)d_in[6];
  const float* Wl2  = (const float*)d_in[7];
  const float* Wr2  = (const float*)d_in[8];
  const float* att2 = (const float*)d_in[9];
  const float* b2   = (const float*)d_in[10];
  const float* dust = (const float*)d_in[11];

  char* p = (char*)d_ws;
  auto alloc = [&](size_t bytes) {
    void* r = (void*)p; p += (bytes + 255) & ~(size_t)255; return r;
  };
  float* xpad   = (float*)alloc(sizeof(float) * NN * D1);   // x zero-padded to K=64
  float* wlp    = (float*)alloc(sizeof(float) * D1 * D1);   // Wl1 padded
  float* wrp    = (float*)alloc(sizeof(float) * D1 * D1);   // Wr1 padded
  float* xl     = (float*)alloc(sizeof(float) * NN * D1);
  float* xr     = (float*)alloc(sizeof(float) * NN * D1);
  float* h1     = (float*)alloc(sizeof(float) * NN * D1);
  float* h2     = (float*)alloc(sizeof(float) * NN * D1);
  float* logits = (float*)alloc(sizeof(float) * (size_t)NET * HH);
  float* mx     = (float*)alloc(sizeof(float) * NN * HH);
  float* denom  = (float*)alloc(sizeof(float) * NN * HH);
  float* norms  = (float*)alloc(sizeof(float) * NN);
  float* Z0     = (float*)alloc(sizeof(float) * (size_t)ZP * ZP);
  float* Z0T    = (float*)alloc(sizeof(float) * (size_t)ZP * ZP);
  float* u      = (float*)alloc(sizeof(float) * ZP);
  float* v      = (float*)alloc(sizeof(float) * ZP);
  unsigned* bar = (unsigned*)alloc(sizeof(unsigned) * 8);
  int* ind0     = (int*)alloc(sizeof(int) * MSZ);
  int* ind1     = (int*)alloc(sizeof(int) * MSZ);
  int* mut0     = (int*)alloc(sizeof(int) * MSZ);
  float* outF   = (float*)d_out;

  // pad layer-1 operands so every GEMM is the K=64 fast path
  k_pad33<<<(NN * D1 + 255) / 256, 256, 0, stream>>>(x,   xpad, NN);
  k_pad33<<<(D1 * D1 + 255) / 256, 256, 0, stream>>>(Wl1, wlp,  D1);
  k_pad33<<<(D1 * D1 + 255) / 256, 256, 0, stream>>>(Wr1, wrp,  D1);

  // ---- two GATv2 layers ----
  for (int layer = 0; layer < 2; ++layer) {
    const float* in   = layer ? h1  : xpad;
    const float* Wl   = layer ? Wl2 : wlp;
    const float* Wr   = layer ? Wr2 : wrp;
    const float* att  = layer ? att2 : att1;
    const float* bias = layer ? b2  : b1;
    const int*   ei   = layer ? ei2 : ei1;
    float*       out  = layer ? h2  : h1;

    k_init_layer <<<(NN * HH + 255) / 256, 256, 0, stream>>>(mx, denom);
    k_gemm64     <<<dim3(NN / 16, D1 / 16), 32, 0, stream>>>(in, Wl, xl, D1);
    k_gemm64     <<<dim3(NN / 16, D1 / 16), 32, 0, stream>>>(in, Wr, xr, D1);
    k_edge_logits<<<(NET + 255) / 256, 256, 0, stream>>>(ei, xl, xr, att, logits, mx);
    k_fix_mx     <<<(NN * HH + 255) / 256, 256, 0, stream>>>(mx);
    k_edge_exp   <<<(NET + 255) / 256, 256, 0, stream>>>(ei, logits, mx, denom);
    k_init_out   <<<(NN * D1 + 255) / 256, 256, 0, stream>>>(out, bias);
    k_aggregate  <<<(NET + 255) / 256, 256, 0, stream>>>(ei, logits, denom, xl, out);
  }

  // ---- log_softmax, norms, cosine-score GEMM into padded Z0 interior ----
  k_logsoftmax<<<NN, 64, 0, stream>>>(h2);
  k_rownorm   <<<NN, 64, 0, stream>>>(h2, norms);
  k_gemm64    <<<dim3(MSZ / 16, MSZ / 16), 32, 0, stream>>>(
      h2, h2 + (size_t)MSZ * D1, Z0, ZP);
  k_finalizeZ0<<<dim3((ZP + 255) / 256, ZP), 256, 0, stream>>>(Z0, Z0T, norms, dust);

  // ---- persistent Sinkhorn (1000 iters, Z0/Z0T L2-resident, float4 loads) ----
  k_init_sink<<<(ZP + 255) / 256, 256, 0, stream>>>(u, v, bar);
  k_sinkhorn <<<SINK_BLOCKS, 256, 0, stream>>>(Z0, Z0T, u, v, bar);

  // ---- outputs: Z, then mutual-match indices ----
  k_writeZ<<<(unsigned)(((size_t)ZN * ZN + 255) / 256), 256, 0, stream>>>(Z0, u, v, outF);
  k_argmax_rows<<<MSZ, 256, 0, stream>>>(Z0, v, ind0);
  k_argmax_cols<<<MSZ, 256, 0, stream>>>(Z0T, u, ind1);
  k_mutual0<<<(MSZ + 255) / 256, 256, 0, stream>>>(ind0, ind1, mut0,
                                                   outF + (size_t)ZN * ZN);
  k_i1     <<<(MSZ + 255) / 256, 256, 0, stream>>>(ind0, ind1, mut0,
                                                   outF + (size_t)ZN * ZN + MSZ);
}
